// SelfAttention_68487548502351
// MI455X (gfx1250) — matrix operand order repaired and checked by reading, not compiled, not run on hardware
//
#include <hip/hip_runtime.h>

// ---------------------------------------------------------------------------
// MI455X / gfx1250 causal self-attention block, bf16 WMMA pipeline.
//   qkv = x @ qkv_w^T + qkv_b        (bf16 WMMA GEMM, scatter to [3,B,H,T,D])
//   flash attention per (b,h)        (bf16 WMMA QK^T and PV, f32 online softmax)
//   out = attn @ out_w^T + out_b     (bf16 WMMA GEMM, fp32 output)
// All matrix math uses v_wmma_f32_16x16x32_bf16 (wave32, 16x16 tiles).
// Staging uses a register->LDS double-buffer pipeline: global loads for tile
// i+1 issue immediately after the barrier for tile i, so their completion wait
// lands after a full WMMA compute phase (one barrier per iteration).
// ---------------------------------------------------------------------------

typedef __bf16          bf16x16 __attribute__((ext_vector_type(16)));
typedef float           f32x8   __attribute__((ext_vector_type(8)));
typedef unsigned short  u16x8   __attribute__((ext_vector_type(8)));

union ABFrag {
    bf16x16 v;
    u16x8   h[2];
};

__device__ __forceinline__ unsigned short f2bf(float f) {
    unsigned int u = __float_as_uint(f);
    u = (u + 0x7FFFu + ((u >> 16) & 1u)) >> 16;   // round-to-nearest-even
    return (unsigned short)u;
}

__device__ __forceinline__ f32x8 wmma_bf16(const ABFrag& a, const ABFrag& b, f32x8 c) {
    // (neg_a, A, neg_b, B, c_mod, C, reuse_a, reuse_b)
    return __builtin_amdgcn_wmma_f32_16x16x32_bf16(false, a.v, false, b.v,
                                                   (short)0, c, false, false);
}

// ---------------------------------------------------------------------------
// f32 -> bf16 conversion (grid-stride)
// ---------------------------------------------------------------------------
__global__ __launch_bounds__(256) void cvt_f32_bf16(const float* __restrict__ in,
                                                    unsigned short* __restrict__ out,
                                                    int n) {
    int i = blockIdx.x * 256 + threadIdx.x;
    int stride = gridDim.x * 256;
    for (; i < n; i += stride) out[i] = f2bf(in[i]);
}

// ---------------------------------------------------------------------------
// GEMM: C[M,N] = A[M,K] @ W[N,K]^T + bias[N]
// Block tile 128x128, 256 threads = 8 waves in 4x2 grid, wave tile 32x64.
// EPI==0: write fp32 to Cf (N==1024).  EPI==1: scatter bf16 qkv to [3,B,H,T,D].
// ---------------------------------------------------------------------------
template<int EPI>
__global__ __launch_bounds__(256) void gemm_bf16_128x128(
    const unsigned short* __restrict__ A,
    const unsigned short* __restrict__ Bw,
    const float* __restrict__ bias,
    float* __restrict__ Cf,
    unsigned short* __restrict__ Cq,
    int K)
{
    constexpr int KS = 40;  // padded LDS row stride (80B = 20 banks, conflict-free)
    __shared__ __align__(16) unsigned short As[2][128 * KS];
    __shared__ __align__(16) unsigned short Bs[2][128 * KS];

    const int tid = threadIdx.x;
    const int w = tid >> 5, l = tid & 31;
    const int wr = w >> 1, wc = w & 1;           // wave grid 4x2
    const int ln = l & 15, hi = l >> 4;
    const int kbase = hi * 8;                    // A-fragment K base per lane-half
    const int koff  = hi * 16;                   // unused; B fragments also read at kbase
    const long mblock = (long)blockIdx.y * 128;
    const long nblock = (long)blockIdx.x * 128;

    const f32x8 fz = {0.f,0.f,0.f,0.f,0.f,0.f,0.f,0.f};
    f32x8 acc[2][4];
    #pragma unroll
    for (int mi = 0; mi < 2; ++mi)
        #pragma unroll
        for (int ni = 0; ni < 4; ++ni) acc[mi][ni] = fz;

    // Cooperative staging: thread -> (row 0..127, K-half 0/16)
    const int srow = tid >> 1;
    const int scol = (tid & 1) * 16;
    const unsigned short* Ag = A  + (mblock + srow) * (long)K + scol;
    const unsigned short* Bg = Bw + (nblock + srow) * (long)K + scol;

    // Preload tile 0 into registers.
    u16x8 a0 = *(const u16x8*)(Ag);
    u16x8 a1 = *(const u16x8*)(Ag + 8);
    u16x8 b0 = *(const u16x8*)(Bg);
    u16x8 b1 = *(const u16x8*)(Bg + 8);

    for (int k0 = 0; k0 < K; k0 += 32) {
        const int buf = (k0 >> 5) & 1;
        *(u16x8*)&As[buf][srow * KS + scol]     = a0;
        *(u16x8*)&As[buf][srow * KS + scol + 8] = a1;
        *(u16x8*)&Bs[buf][srow * KS + scol]     = b0;
        *(u16x8*)&Bs[buf][srow * KS + scol + 8] = b1;
        __syncthreads();
        if (k0 + 32 < K) {      // issue next tile's global loads before compute
            a0 = *(const u16x8*)(Ag + k0 + 32);
            a1 = *(const u16x8*)(Ag + k0 + 40);
            b0 = *(const u16x8*)(Bg + k0 + 32);
            b1 = *(const u16x8*)(Bg + k0 + 40);
            if (k0 + 64 < K) {  // L2 warm for the tile after next
                __builtin_prefetch(Ag + k0 + 64, 0, 0);
                __builtin_prefetch(Bg + k0 + 64, 0, 0);
            }
        }

        ABFrag af[2], bf_[4];
        #pragma unroll
        for (int mi = 0; mi < 2; ++mi) {
            const unsigned short* p = &As[buf][(wr * 32 + mi * 16 + ln) * KS + kbase];
            af[mi].h[0] = *(const u16x8*)p;         // K = kbase..kbase+7
            af[mi].h[1] = *(const u16x8*)(p + 16);  // K = 16+kbase..
        }
        #pragma unroll
        for (int ni = 0; ni < 4; ++ni) {
            const unsigned short* p = &Bs[buf][(wc * 64 + ni * 16 + ln) * KS + kbase];
            bf_[ni].h[0] = *(const u16x8*)p;        // K = kbase..kbase+7
            bf_[ni].h[1] = *(const u16x8*)(p + 16); // K = 16+kbase..
        }
        #pragma unroll
        for (int mi = 0; mi < 2; ++mi)
            #pragma unroll
            for (int ni = 0; ni < 4; ++ni)
                acc[mi][ni] = wmma_bf16(af[mi], bf_[ni], acc[mi][ni]);
    }

    // Epilogue.  C/D layout: element r of lane l -> (M = r + 8*(l>>4), N = l&15)
    #pragma unroll
    for (int mi = 0; mi < 2; ++mi)
        #pragma unroll
        for (int ni = 0; ni < 4; ++ni)
            #pragma unroll
            for (int r = 0; r < 8; ++r) {
                long gm = mblock + wr * 32 + mi * 16 + r + 8 * hi;
                long gn = nblock + wc * 64 + ni * 16 + ln;
                float v = acc[mi][ni][r] + bias[gn];
                if (EPI == 0) {
                    Cf[gm * 1024 + gn] = v;
                } else {
                    long which = gn >> 10, c = gn & 1023;
                    long h = c >> 6, d = c & 63;
                    long b = gm >> 11, t = gm & 2047;
                    Cq[(((which * 4 + b) * 16 + h) * 2048 + t) * 64 + d] = f2bf(v);
                }
            }
}

// ---------------------------------------------------------------------------
// Flash attention.  Grid: (T/128 q-blocks, B*H).  256 threads = 8 waves.
// Wave w owns q rows [qbase + 16w, +16).  Key blocks of 64, causal,
// double-buffered K/V staging.
// ---------------------------------------------------------------------------
__global__ __launch_bounds__(256) void flash_attn(
    const unsigned short* __restrict__ qkv,   // [3][B][H][T][D] bf16
    unsigned short* __restrict__ attn_out)    // [B*T][C] bf16, c = h*64 + d
{
    constexpr int  KS = 72;                   // 144B row stride, conflict-free
    constexpr long HSTRIDE = 2048L * 64;
    constexpr long WSTRIDE = 4L * 16 * 2048 * 64;
    __shared__ __align__(16) unsigned short Ks[2][64 * KS];   // K tile  [kcol][d]
    __shared__ __align__(16) unsigned short Vt[2][64 * KS];   // V^T     [d][kcol]
    __shared__ __align__(16) unsigned short Ps[8 * 16 * KS];  // per-wave P tiles

    const int tid = threadIdx.x;
    const int w = tid >> 5, l = tid & 31;
    const int ln = l & 15, hi = l >> 4;
    const int kbase = hi * 8, koff = hi * 16;
    const int bh    = blockIdx.y;
    const int qbase = blockIdx.x * 128;
    const unsigned short* Qp = qkv + (long)bh * HSTRIDE;
    const unsigned short* Kp = Qp + WSTRIDE;
    const unsigned short* Vp = Qp + 2 * WSTRIDE;

    // Q A-fragments for this wave's 16 rows, D split into two K=32 steps.
    const int qrow_w = qbase + w * 16;
    ABFrag qf[2];
    #pragma unroll
    for (int ds = 0; ds < 2; ++ds) {
        const unsigned short* p = Qp + (long)(qrow_w + ln) * 64 + ds * 32 + kbase;
        qf[ds].h[0] = *(const u16x8*)p;
        qf[ds].h[1] = *(const u16x8*)(p + 16);
    }

    const f32x8 fz = {0.f,0.f,0.f,0.f,0.f,0.f,0.f,0.f};
    float mr[8], lr[8];
    f32x8 Oa[4];
    #pragma unroll
    for (int r = 0; r < 8; ++r) { mr[r] = -1e30f; lr[r] = 0.f; }
    #pragma unroll
    for (int ni = 0; ni < 4; ++ni) Oa[ni] = fz;

    const int srow = tid >> 2;            // 0..63
    const int scol = (tid & 3) * 16;      // 0,16,32,48
    const int nkb  = 2 * blockIdx.x + 2;  // causal: key blocks covering <= qbase+127
    const float SC = 0.125f * 1.44269504088896f;   // 1/sqrt(64) * log2(e)

    // Preload K/V tile 0 into registers.
    u16x8 kv0 = *(const u16x8*)(Kp + (long)srow * 64 + scol);
    u16x8 kv1 = *(const u16x8*)(Kp + (long)srow * 64 + scol + 8);
    u16x8 vv0 = *(const u16x8*)(Vp + (long)srow * 64 + scol);
    u16x8 vv1 = *(const u16x8*)(Vp + (long)srow * 64 + scol + 8);

    for (int kcb = 0; kcb < nkb; ++kcb) {
        const int kc0 = kcb * 64;
        const int buf = kcb & 1;
        // Stage tile kcb from registers: K row-major, V transposed.
        *(u16x8*)&Ks[buf][srow * KS + scol]     = kv0;
        *(u16x8*)&Ks[buf][srow * KS + scol + 8] = kv1;
        #pragma unroll
        for (int j = 0; j < 8; ++j) Vt[buf][(scol + j) * KS + srow]     = vv0[j];
        #pragma unroll
        for (int j = 0; j < 8; ++j) Vt[buf][(scol + 8 + j) * KS + srow] = vv1[j];
        __syncthreads();
        if (kcb + 1 < nkb) {   // issue next tile's global loads before compute
            const long roff = (long)(kc0 + 64 + srow) * 64 + scol;
            kv0 = *(const u16x8*)(Kp + roff);
            kv1 = *(const u16x8*)(Kp + roff + 8);
            vv0 = *(const u16x8*)(Vp + roff);
            vv1 = *(const u16x8*)(Vp + roff + 8);
        }

        if (kc0 <= qrow_w + 15) {   // wave-uniform causal skip (barriers stay convergent)
            // ---- S = Q K^T : 4 column tiles x 2 K-steps over D ----
            f32x8 sa[4];
            #pragma unroll
            for (int j = 0; j < 4; ++j) sa[j] = fz;
            #pragma unroll
            for (int j = 0; j < 4; ++j)
                #pragma unroll
                for (int ds = 0; ds < 2; ++ds) {
                    ABFrag kf;
                    const unsigned short* p = &Ks[buf][(j * 16 + ln) * KS + ds * 32 + kbase];
                    kf.h[0] = *(const u16x8*)p;
                    kf.h[1] = *(const u16x8*)(p + 16);
                    sa[j] = wmma_bf16(qf[ds], kf, sa[j]);
                }
            // ---- scale + causal mask (element (m,n): m = r+8*hi, n = ln) ----
            const int qr0 = qrow_w + 8 * hi;
            #pragma unroll
            for (int j = 0; j < 4; ++j)
                #pragma unroll
                for (int r = 0; r < 8; ++r) {
                    float s = sa[j][r] * SC;
                    if (kc0 + j * 16 + ln > qr0 + r) s = -1e30f;
                    sa[j][r] = s;
                }
            // ---- online softmax: row max / rescale / exp / row sum ----
            float fct[8];
            #pragma unroll
            for (int r = 0; r < 8; ++r) {
                float t = fmaxf(fmaxf(sa[0][r], sa[1][r]), fmaxf(sa[2][r], sa[3][r]));
                t = fmaxf(t, __shfl_xor(t, 1));
                t = fmaxf(t, __shfl_xor(t, 2));
                t = fmaxf(t, __shfl_xor(t, 4));
                t = fmaxf(t, __shfl_xor(t, 8));
                float mn = fmaxf(mr[r], t);
                fct[r] = exp2f(mr[r] - mn);
                mr[r] = mn;
            }
            #pragma unroll
            for (int j = 0; j < 4; ++j)
                #pragma unroll
                for (int r = 0; r < 8; ++r)
                    sa[j][r] = exp2f(sa[j][r] - mr[r]);
            #pragma unroll
            for (int r = 0; r < 8; ++r) {
                float s = sa[0][r] + sa[1][r] + sa[2][r] + sa[3][r];
                s += __shfl_xor(s, 1);
                s += __shfl_xor(s, 2);
                s += __shfl_xor(s, 4);
                s += __shfl_xor(s, 8);
                lr[r] = lr[r] * fct[r] + s;
            }
            #pragma unroll
            for (int ni = 0; ni < 4; ++ni)
                #pragma unroll
                for (int r = 0; r < 8; ++r)
                    Oa[ni][r] *= fct[r];
            // ---- P: C-layout -> LDS -> A-layout (per-wave region, no barrier) ----
            #pragma unroll
            for (int j = 0; j < 4; ++j)
                #pragma unroll
                for (int r = 0; r < 8; ++r)
                    Ps[(w * 16 + r + 8 * hi) * KS + j * 16 + ln] = f2bf(sa[j][r]);
            // ---- O += P V : 4 d-tiles x 2 K-steps over kcols ----
            #pragma unroll
            for (int ni = 0; ni < 4; ++ni)
                #pragma unroll
                for (int kc2 = 0; kc2 < 2; ++kc2) {
                    ABFrag pa, vb;
                    const unsigned short* pp = &Ps[(w * 16 + ln) * KS + kc2 * 32 + kbase];
                    pa.h[0] = *(const u16x8*)pp;
                    pa.h[1] = *(const u16x8*)(pp + 16);
                    const unsigned short* vp = &Vt[buf][(ni * 16 + ln) * KS + kc2 * 32 + kbase];
                    vb.h[0] = *(const u16x8*)vp;
                    vb.h[1] = *(const u16x8*)(vp + 16);
                    Oa[ni] = wmma_bf16(pa, vb, Oa[ni]);
                }
        }
    }

    // ---- epilogue: O /= l, store bf16 rows at [b*T + t][h*64 + d] ----
    const long brow = (long)(bh >> 4) * 2048;
    const int h = bh & 15;
    #pragma unroll
    for (int ni = 0; ni < 4; ++ni)
        #pragma unroll
        for (int r = 0; r < 8; ++r) {
            long t = qbase + w * 16 + r + 8 * hi;
            float o = Oa[ni][r] / lr[r];
            attn_out[(brow + t) * 1024 + h * 64 + ni * 16 + ln] = f2bf(o);
        }
}

// ---------------------------------------------------------------------------
// Host launcher
// ---------------------------------------------------------------------------
extern "C" void kernel_launch(void* const* d_in, const int* in_sizes, int n_in,
                              void* d_out, int out_size, void* d_ws, size_t ws_size,
                              hipStream_t stream) {
    (void)in_sizes; (void)n_in; (void)out_size; (void)ws_size;
    const float* x     = (const float*)d_in[0];   // [4,2048,1024]
    const float* qkv_w = (const float*)d_in[1];   // [3072,1024]
    const float* qkv_b = (const float*)d_in[2];   // [3072]
    const float* out_w = (const float*)d_in[3];   // [1024,1024]
    const float* out_b = (const float*)d_in[4];   // [1024]
    float* out = (float*)d_out;                   // [4,2048,1024] f32

    const long NX   = 8192L * 1024;
    const long NQW  = 3072L * 1024;
    const long NOW  = 1024L * 1024;
    const long NQKV = 3L * 4 * 16 * 2048 * 64;
    unsigned short* xb   = (unsigned short*)d_ws;   // ~92 MB total workspace
    unsigned short* wqb  = xb   + NX;
    unsigned short* wob  = wqb  + NQW;
    unsigned short* qkvb = wob  + NOW;
    unsigned short* attb = qkvb + NQKV;

    cvt_f32_bf16<<<2048, 256, 0, stream>>>(x,     xb,  (int)NX);
    cvt_f32_bf16<<<1024, 256, 0, stream>>>(qkv_w, wqb, (int)NQW);
    cvt_f32_bf16<<<512,  256, 0, stream>>>(out_w, wob, (int)NOW);

    // qkv = x @ qkv_w^T + qkv_b  -> scatter bf16 to [3,B,H,T,D]
    gemm_bf16_128x128<1><<<dim3(24, 64), 256, 0, stream>>>(xb, wqb, qkv_b,
                                                           nullptr, qkvb, 1024);
    // causal flash attention -> bf16 [B*T, C]
    flash_attn<<<dim3(16, 64), 256, 0, stream>>>(qkvb, attb);
    // out = attn @ out_w^T + out_b -> fp32
    gemm_bf16_128x128<0><<<dim3(8, 64), 256, 0, stream>>>(attb, wob, out_b,
                                                          out, nullptr, 1024);
}
